// EinFFT_29695403884950
// MI455X (gfx1250) — matrix operand order repaired and checked by reading, not compiled, not run on hardware
//
#include <hip/hip_runtime.h>
#include <hip/hip_bf16.h>

// ---------------------------------------------------------------------------
// EinFFT for MI455X (gfx1250, wave32, WMMA).
// Pipeline: ln_stats -> fft_fwd(+LN fused) -> bf16 WMMA GEMM x2 -> ifft(+residual)
// ---------------------------------------------------------------------------

typedef unsigned short u16_t;
typedef __attribute__((ext_vector_type(16))) __bf16 v16bf;
typedef __attribute__((ext_vector_type(8)))  float  v8f;

#define DIM      768
#define NB       4
#define BS       192
#define BATCH    64
#define NPIX     1024
#define BN_TOTAL (BATCH * NPIX)      // 65536 columns for the GEMMs
#define LAM      0.01f
#define ORTHO    (1.0f / 64.0f)      // 1/sqrt(1024*4), each FFT direction
#define FPAD     20                  // padded fragment lane stride (u16 units)

union FragU {
    v16bf v;
    u16_t s[16];
    uint2 d2[4];
};
union V16 {
    uint4 v;
    u16_t s[8];
};

__device__ __forceinline__ u16_t f2bf(float f) {
    unsigned u = __float_as_uint(f);
    unsigned r = u + 0x7FFFu + ((u >> 16) & 1u);   // round-to-nearest-even
    return (u16_t)(r >> 16);
}
__device__ __forceinline__ float bf2f(u16_t h) {
    return __uint_as_float(((unsigned)h) << 16);
}

__device__ __forceinline__ float2 c_add(float2 a, float2 b) { return make_float2(a.x + b.x, a.y + b.y); }
__device__ __forceinline__ float2 c_sub(float2 a, float2 b) { return make_float2(a.x - b.x, a.y - b.y); }
__device__ __forceinline__ float2 c_muli (float2 a) { return make_float2(-a.y,  a.x); }  //  i*a
__device__ __forceinline__ float2 c_mulni(float2 a) { return make_float2( a.y, -a.x); }  // -i*a

// Read a 16-u16 fragment stored contiguously (padded stride) with 4x ds_load_b64.
__device__ __forceinline__ void ld_frag(FragU& f, const u16_t* p) {
    f.d2[0] = *(const uint2*)(p + 0);
    f.d2[1] = *(const uint2*)(p + 4);
    f.d2[2] = *(const uint2*)(p + 8);
    f.d2[3] = *(const uint2*)(p + 12);
}

// ---------------------------------------------------------------------------
// Kernel 0: weight prep. wT[(L*3+type)*4+q][k][d] bf16, type 0 = w_r, 1 = w_i,
// 2 = -w_i.  d contiguous so WMMA A-fragments are two aligned 16B loads.
// ---------------------------------------------------------------------------
__global__ __launch_bounds__(256) void prep_w_kernel(const float* __restrict__ w1,
                                                     const float* __restrict__ w2,
                                                     u16_t* __restrict__ wt) {
    int idx = blockIdx.x * 256 + threadIdx.x;       // over 2*4*192*192
    if (idx >= 2 * 4 * BS * BS) return;
    int L  = idx / (4 * BS * BS);
    int r  = idx % (4 * BS * BS);
    int q  = r / (BS * BS);
    int rr = r % (BS * BS);
    int d  = rr / BS;
    int k  = rr % BS;
    const float* w = L ? w2 : w1;                   // (2, 4, 192, 192): [t][q][d][k]
    float vr = w[((size_t)(0 * 4 + q) * BS + d) * BS + k];
    float vi = w[((size_t)(4 + q) * BS + d) * BS + k];
    size_t o = (size_t)k * BS + d;
    wt[(size_t)(L * 12 + 0 * 4 + q) * (BS * BS) + o] = f2bf(vr);
    wt[(size_t)(L * 12 + 1 * 4 + q) * (BS * BS) + o] = f2bf(vi);
    wt[(size_t)(L * 12 + 2 * 4 + q) * (BS * BS) + o] = f2bf(-vi);
}

// ---------------------------------------------------------------------------
// Kernel 1: LayerNorm statistics per (b, n).  Coalesced: a wave reads 32
// consecutive n for each channel c (4KB stride between channels), with
// software prefetch (global_prefetch_b8) to cover the strided walk.
// ---------------------------------------------------------------------------
__global__ __launch_bounds__(256) void ln_stats_kernel(const float* __restrict__ x,
                                                       float* __restrict__ mu,
                                                       float* __restrict__ rstd) {
    int gid = blockIdx.x * 256 + threadIdx.x;        // 0 .. 65535
    int b = gid >> 10, n = gid & 1023;
    const float* p = x + ((size_t)b * DIM << 10) + n;
    float s = 0.f, s2 = 0.f;
    for (int c = 0; c < DIM; c += 4) {
        __builtin_prefetch(p + ((size_t)(c + 16) << 10), 0, 1);
        #pragma unroll
        for (int u = 0; u < 4; ++u) {
            float v = p[(size_t)(c + u) << 10];
            s += v; s2 += v * v;
        }
    }
    float m = s * (1.f / DIM);
    float var = s2 * (1.f / DIM) - m * m;
    mu[gid] = m;
    rstd[gid] = rsqrtf(var + 1e-5f);
}

// ---------------------------------------------------------------------------
// In-LDS radix-2 DIT FFT, length 1024, 4 rows in parallel, 256 threads.
// Input must be stored bit-reversed.  sign = -1 forward, +1 inverse (unscaled).
// ---------------------------------------------------------------------------
__device__ void fft1024(float2 (*buf)[1024], int tid, float sign) {
    for (int st = 1; st <= 10; ++st) {
        int half = 1 << (st - 1);
        float w0 = sign * 6.28318530717958647692f / (float)(half << 1);
        __syncthreads();
        for (int idx = tid; idx < 4 * 512; idx += 256) {
            int row = idx >> 9;
            int k   = idx & 511;
            int j   = k & (half - 1);
            int p0  = ((k >> (st - 1)) << st) + j;
            int p1  = p0 + half;
            float sn, cs;
            __sincosf(w0 * (float)j, &sn, &cs);
            float2 a = buf[row][p0], bb = buf[row][p1];
            float2 t = make_float2(bb.x * cs - bb.y * sn, bb.x * sn + bb.y * cs);
            buf[row][p0] = make_float2(a.x + t.x, a.y + t.y);
            buf[row][p1] = make_float2(a.x - t.x, a.y - t.y);
        }
    }
    __syncthreads();
}

// ---------------------------------------------------------------------------
// Kernel 2: fused LN + FFT(N=1024) + length-4 q-mix + ortho scale, bf16 out.
// One workgroup per (b, d).  Output layout: X[(q*192+d)][b*1024+n] bf16.
// ---------------------------------------------------------------------------
__global__ __launch_bounds__(256) void fft_fwd_kernel(const float* __restrict__ x,
                                                      const float* __restrict__ mu,
                                                      const float* __restrict__ rstd,
                                                      const float* __restrict__ gamma,
                                                      const float* __restrict__ beta,
                                                      u16_t* __restrict__ Xr,
                                                      u16_t* __restrict__ Xi) {
    __shared__ float2 buf[4][1024];
    const int tid = threadIdx.x;
    const int b = blockIdx.x / BS;
    const int d = blockIdx.x % BS;
    const float* mup = mu   + ((size_t)b << 10);
    const float* rsp = rstd + ((size_t)b << 10);
    #pragma unroll
    for (int q = 0; q < 4; ++q) {
        const int c = q * BS + d;
        const float g = gamma[c], be = beta[c];
        const float* row = x + (((size_t)b * DIM + c) << 10);
        for (int i = tid; i < 1024; i += 256) {
            float v = (row[i] - mup[i]) * rsp[i] * g + be;
            buf[q][__brev((unsigned)i) >> 22] = make_float2(v, 0.f);
        }
    }
    fft1024(buf, tid, -1.f);
    for (int i = tid; i < 1024; i += 256) {
        float2 X0 = buf[0][i], X1 = buf[1][i], X2 = buf[2][i], X3 = buf[3][i];
        float2 E  = c_add(X0, X2), O  = c_add(X1, X3);
        float2 Em = c_sub(X0, X2), Om = c_sub(X1, X3);
        float2 Y[4];
        Y[0] = c_add(E, O);
        Y[2] = c_sub(E, O);
        Y[1] = c_add(Em, c_mulni(Om));   // X0 - iX1 - X2 + iX3
        Y[3] = c_add(Em, c_muli(Om));
        #pragma unroll
        for (int q = 0; q < 4; ++q) {
            const size_t off = ((size_t)(q * BS + d) << 16) + ((size_t)b << 10) + i;
            Xr[off] = f2bf(Y[q].x * ORTHO);
            Xi[off] = f2bf(Y[q].y * ORTHO);
        }
    }
}

// ---------------------------------------------------------------------------
// Kernels 3 & 4: complex block-diagonal GEMM, bf16 WMMA, f32 accumulate.
//   outR[k,bn] = act( sum_d wr[k,d]*Xr[d,bn] - wi[k,d]*Xi[d,bn] + biasR[k] )
//   outI[k,bn] = act( sum_d wi[k,d]*Xr[d,bn] + wr[k,d]*Xi[d,bn] + biasI[k] )
// Workgroup: (q, 32-column bn tile); 8 waves cover the 12 M-tiles (192 rows).
// The 32x32 X tiles are scattered into LDS *in WMMA B-fragment order*
// ([cs][lane][e], lane stride 20 u16 -> conflict-free), so every wave builds a
// fragment with 4 aligned ds_load_b64 and zero packing VALU.  Global tile
// loads are double-buffered in registers across the barrier.
// ---------------------------------------------------------------------------
__global__ __launch_bounds__(256) void gemm_kernel(const u16_t* __restrict__ wt,
                                                   int layerOff12,
                                                   const float* __restrict__ bias,
                                                   const u16_t* __restrict__ inR,
                                                   const u16_t* __restrict__ inI,
                                                   u16_t* __restrict__ outR,
                                                   u16_t* __restrict__ outI,
                                                   int act) {
    __shared__ u16_t fragR[2][32][FPAD];
    __shared__ u16_t fragI[2][32][FPAD];
    __shared__ float biasLds[2][BS];

    const int tid  = threadIdx.x;
    const int wave = tid >> 5;
    const int lane = tid & 31;
    const int q    = blockIdx.y;
    const int bn0  = blockIdx.x * 32;

    if (tid < BS) {
        biasLds[0][tid] = bias[q * BS + tid];        // b[0][q][k]
        biasLds[1][tid] = bias[(4 + q) * BS + tid];  // b[1][q][k]
    }

    v8f accR[2][2], accI[2][2];
    #pragma unroll
    for (int s = 0; s < 2; ++s)
        #pragma unroll
        for (int cs = 0; cs < 2; ++cs) { accR[s][cs] = {}; accI[s][cs] = {}; }

    // cooperative staging assignment: threads 0-127 -> R tile, 128-255 -> I
    const int tile = tid >> 7;
    const int t7   = tid & 127;
    const int kk   = t7 >> 2;         // K row 0..31 within chunk
    const int seg  = t7 & 3;          // 8-column segment
    u16_t (*frag)[32][FPAD] = tile ? fragI : fragR;
    const u16_t* srcBase = (tile ? inI : inR) +
                           ((size_t)(q * BS + kk) << 16) + bn0 + seg * 8;

    uint4 gv = *(const uint4*)(srcBase);             // prefetch chunk kc=0

    for (int kc = 0; kc < BS; kc += 32) {
        __syncthreads();                             // LDS consumers done
        {   // scatter 8 u16 into fragment order
            V16 u; u.v = gv;
            #pragma unroll
            for (int j = 0; j < 8; ++j) {
                const int col = seg * 8 + j;
                frag[col >> 4][(col & 15) | ((kk & 8) << 1)][(kk & 7) | ((kk & 16) >> 1)] = u.s[j];
            }
        }
        __syncthreads();
        if (kc + 32 < BS)                            // overlap next global load
            gv = *(const uint4*)(srcBase + ((size_t)(kc + 32) << 16));

        // B fragments: 4x ds_load_b64 each, no packing.
        FragU br0, br1, bi0, bi1;
        ld_frag(br0, &fragR[0][lane][0]);
        ld_frag(br1, &fragR[1][lane][0]);
        ld_frag(bi0, &fragI[0][lane][0]);
        ld_frag(bi1, &fragI[1][lane][0]);

        #pragma unroll
        for (int s = 0; s < 2; ++s) {
            const int mt = wave + 8 * s;
            if (mt >= 12) break;
            const int m  = mt * 16 + (lane & 15);
            const int o1 = kc + ((lane & 16) ? 8 : 0);   // documented 16-bit A layout
            const u16_t* rowR = wt + (size_t)(layerOff12 + 0 * 4 + q) * (BS * BS) + (size_t)m * BS;
            const u16_t* rowI = wt + (size_t)(layerOff12 + 1 * 4 + q) * (BS * BS) + (size_t)m * BS;
            const u16_t* rowN = wt + (size_t)(layerOff12 + 2 * 4 + q) * (BS * BS) + (size_t)m * BS;
            FragU ar, ai, an;
            ar.d2[0] = ((const uint2*)(rowR + o1))[0];      ar.d2[1] = ((const uint2*)(rowR + o1))[1];
            ar.d2[2] = ((const uint2*)(rowR + o1 + 16))[0]; ar.d2[3] = ((const uint2*)(rowR + o1 + 16))[1];
            ai.d2[0] = ((const uint2*)(rowI + o1))[0];      ai.d2[1] = ((const uint2*)(rowI + o1))[1];
            ai.d2[2] = ((const uint2*)(rowI + o1 + 16))[0]; ai.d2[3] = ((const uint2*)(rowI + o1 + 16))[1];
            an.d2[0] = ((const uint2*)(rowN + o1))[0];      an.d2[1] = ((const uint2*)(rowN + o1))[1];
            an.d2[2] = ((const uint2*)(rowN + o1 + 16))[0]; an.d2[3] = ((const uint2*)(rowN + o1 + 16))[1];

            // 4 independent accumulators first, then their second terms:
            // every D->C chain has distance >= 4.
            accR[s][0] = __builtin_amdgcn_wmma_f32_16x16x32_bf16(false, ar.v, false, br0.v, (short)0, accR[s][0], false, false);
            accR[s][1] = __builtin_amdgcn_wmma_f32_16x16x32_bf16(false, ar.v, false, br1.v, (short)0, accR[s][1], false, false);
            accI[s][0] = __builtin_amdgcn_wmma_f32_16x16x32_bf16(false, ai.v, false, br0.v, (short)0, accI[s][0], false, false);
            accI[s][1] = __builtin_amdgcn_wmma_f32_16x16x32_bf16(false, ai.v, false, br1.v, (short)0, accI[s][1], false, false);
            accR[s][0] = __builtin_amdgcn_wmma_f32_16x16x32_bf16(false, an.v, false, bi0.v, (short)0, accR[s][0], false, false);
            accR[s][1] = __builtin_amdgcn_wmma_f32_16x16x32_bf16(false, an.v, false, bi1.v, (short)0, accR[s][1], false, false);
            accI[s][0] = __builtin_amdgcn_wmma_f32_16x16x32_bf16(false, ar.v, false, bi0.v, (short)0, accI[s][0], false, false);
            accI[s][1] = __builtin_amdgcn_wmma_f32_16x16x32_bf16(false, ar.v, false, bi1.v, (short)0, accI[s][1], false, false);
        }
    }

    // Epilogue: bias + activation, bf16 store.  C layout: VGPR r -> row r (+8 hi half).
    #pragma unroll
    for (int s = 0; s < 2; ++s) {
        const int mt = wave + 8 * s;
        if (mt >= 12) break;
        #pragma unroll
        for (int cs = 0; cs < 2; ++cs) {
            #pragma unroll
            for (int r = 0; r < 8; ++r) {
                const int m = mt * 16 + r + ((lane & 16) ? 8 : 0);
                float vR = accR[s][cs][r] + biasLds[0][m];
                float vI = accI[s][cs][r] + biasLds[1][m];
                if (act == 0) {                     // ReLU
                    vR = fmaxf(vR, 0.f);
                    vI = fmaxf(vI, 0.f);
                } else {                            // softshrink
                    vR = (vR > LAM) ? vR - LAM : ((vR < -LAM) ? vR + LAM : 0.f);
                    vI = (vI > LAM) ? vI - LAM : ((vI < -LAM) ? vI + LAM : 0.f);
                }
                const size_t off = ((size_t)(q * BS + m) << 16) + bn0 + cs * 16 + (lane & 15);
                outR[off] = f2bf(vR);
                outI[off] = f2bf(vI);
            }
        }
    }
}

// ---------------------------------------------------------------------------
// Kernel 5: inverse q-mix + inverse FFT (sign +1) + ortho scale + residual.
// ---------------------------------------------------------------------------
__global__ __launch_bounds__(256) void ifft_kernel(const u16_t* __restrict__ Sr,
                                                   const u16_t* __restrict__ Si,
                                                   const float* __restrict__ x,
                                                   float* __restrict__ out) {
    __shared__ float2 buf[4][1024];
    const int tid = threadIdx.x;
    const int b = blockIdx.x / BS;
    const int d = blockIdx.x % BS;
    for (int i = tid; i < 1024; i += 256) {
        float2 Y[4];
        #pragma unroll
        for (int q = 0; q < 4; ++q) {
            const size_t off = ((size_t)(q * BS + d) << 16) + ((size_t)b << 10) + i;
            Y[q] = make_float2(bf2f(Sr[off]), bf2f(Si[off]));
        }
        float2 E  = c_add(Y[0], Y[2]), O  = c_add(Y[1], Y[3]);
        float2 Em = c_sub(Y[0], Y[2]), Om = c_sub(Y[1], Y[3]);
        const unsigned ir = __brev((unsigned)i) >> 22;
        buf[0][ir] = c_add(E, O);
        buf[2][ir] = c_sub(E, O);
        buf[1][ir] = c_add(Em, c_muli(Om));    // Y0 + iY1 - Y2 - iY3
        buf[3][ir] = c_add(Em, c_mulni(Om));
    }
    fft1024(buf, tid, +1.f);
    #pragma unroll
    for (int q = 0; q < 4; ++q) {
        const int c = q * BS + d;
        const size_t ob = ((size_t)b * DIM + c) << 10;
        for (int i = tid; i < 1024; i += 256) {
            out[ob + i] = buf[q][i].x * ORTHO + x[ob + i];   // real part + residual
        }
    }
}

// ---------------------------------------------------------------------------
extern "C" void kernel_launch(void* const* d_in, const int* in_sizes, int n_in,
                              void* d_out, int out_size, void* d_ws, size_t ws_size,
                              hipStream_t stream) {
    (void)in_sizes; (void)n_in; (void)out_size; (void)ws_size;
    const float* x     = (const float*)d_in[0];
    const float* w1    = (const float*)d_in[1];
    const float* w2    = (const float*)d_in[2];
    const float* b1    = (const float*)d_in[3];
    const float* b2    = (const float*)d_in[4];
    const float* gamma = (const float*)d_in[5];
    const float* beta  = (const float*)d_in[6];
    float* out = (float*)d_out;

    char* ws = (char*)d_ws;
    const size_t WT_BYTES   = (size_t)2 * 3 * 4 * BS * BS * 2;   // 1,769,472
    const size_t MU_BYTES   = (size_t)BN_TOTAL * 4;              // 262,144
    const size_t SPEC_BYTES = (size_t)DIM * BN_TOTAL * 2;        // 100,663,296 (bf16)

    u16_t* wt   = (u16_t*)(ws);
    float* mu   = (float*)(ws + WT_BYTES);
    float* rs   = (float*)(ws + WT_BYTES + MU_BYTES);
    char*  spec = ws + WT_BYTES + 2 * MU_BYTES;                  // 256B aligned
    u16_t* Xr = (u16_t*)(spec);
    u16_t* Xi = (u16_t*)(spec + SPEC_BYTES);
    u16_t* R1 = (u16_t*)(spec + 2 * SPEC_BYTES);
    u16_t* I1 = (u16_t*)(spec + 3 * SPEC_BYTES);
    u16_t* SrP = Xr;   // layer-2 output reuses Xr/Xi (dead after layer 1)
    u16_t* SiP = Xi;

    prep_w_kernel<<<(2 * 4 * BS * BS) / 256, 256, 0, stream>>>(w1, w2, wt);
    ln_stats_kernel<<<BN_TOTAL / 256, 256, 0, stream>>>(x, mu, rs);
    fft_fwd_kernel<<<BATCH * BS, 256, 0, stream>>>(x, mu, rs, gamma, beta, Xr, Xi);

    dim3 g(BN_TOTAL / 32, 4, 1);
    gemm_kernel<<<g, 256, 0, stream>>>(wt, 0,  b1, Xr, Xi, R1, I1, /*act=*/0);
    gemm_kernel<<<g, 256, 0, stream>>>(wt, 12, b2, R1, I1, SrP, SiP, /*act=*/1);

    ifft_kernel<<<BATCH * BS, 256, 0, stream>>>(SrP, SiP, x, out);
}
